// AdaptiveNeighbourSampling_23381801959629
// MI455X (gfx1250) — matrix operand order repaired and checked by reading, not compiled, not run on hardware
//
#include <hip/hip_runtime.h>
#include <stdint.h>

// ---------------------------------------------------------------------------
// AdaptiveNeighbourSampling on MI455X (gfx1250, wave32)
//   N=8192 nodes, D=128 features, K=16 neighbors
// Fused: bf16 WMMA similarity GEMM + adjacency weighting + masked top-k,
// single streaming pass over the 256 MB adjacency (HBM-bound, ~11 us floor).
// ---------------------------------------------------------------------------

#define N_NODES 8192
#define DIMS    128
#define KSEL    16

typedef __attribute__((ext_vector_type(16))) __bf16 v16bf;
typedef __attribute__((ext_vector_type(8)))  float  v8f;

struct B32x8 { uint4 a, b; };

static __device__ __forceinline__ v16bf as_v16bf(uint4 a, uint4 b) {
  B32x8 t{a, b};
  return __builtin_bit_cast(v16bf, t);
}

// float -> bf16 round-to-nearest-even (inputs are finite)
static __device__ __forceinline__ unsigned short f2bf(float f) {
  uint32_t u = __builtin_bit_cast(uint32_t, f);
  u += 0x7FFFu + ((u >> 16) & 1u);
  return (unsigned short)(u >> 16);
}

// Sorted-descending 16-entry insert (fully unrolled; no dynamic reg indexing).
static __device__ __forceinline__ void ins_desc(float (&v)[KSEL], int (&id)[KSEL],
                                                float nv, int ni) {
#pragma unroll
  for (int i = KSEL - 1; i >= 0; --i) {
    float pv = (i > 0) ? v[i - 1] : __builtin_inff();
    int   pi = (i > 0) ? id[i - 1] : -1;
    bool  up   = nv > pv;     // candidate belongs above slot i-1 -> shift down
    bool  here = nv > v[i];   // candidate lands here
    float cv = here ? nv : v[i];
    int   ci = here ? ni : id[i];
    v[i]  = up ? pv : cv;
    id[i] = up ? pi : ci;
  }
}

// Sorted-ascending 16-entry insert (keeps the 16 smallest).
static __device__ __forceinline__ void ins_asc(float (&v)[KSEL], int (&id)[KSEL],
                                               float nv, int ni) {
#pragma unroll
  for (int i = KSEL - 1; i >= 0; --i) {
    float pv = (i > 0) ? v[i - 1] : -__builtin_inff();
    int   pi = (i > 0) ? id[i - 1] : -1;
    bool  up   = nv < pv;
    bool  here = nv < v[i];
    float cv = here ? nv : v[i];
    int   ci = here ? ni : id[i];
    v[i]  = up ? pv : cv;
    id[i] = up ? pi : ci;
  }
}

// --------------------------- Kernel 1: normalize ---------------------------
// One wave per row: L2-normalize (eps=1e-12 like F.normalize) and emit bf16.
__global__ void __launch_bounds__(256)
l2norm_bf16_kernel(const float* __restrict__ x, unsigned short* __restrict__ xn) {
  const int row  = (blockIdx.x * blockDim.x + threadIdx.x) >> 5;
  const int lane = threadIdx.x & 31;
  const float4 v = reinterpret_cast<const float4*>(x)[row * 32 + lane];
  float ss = v.x * v.x + v.y * v.y + v.z * v.z + v.w * v.w;
#pragma unroll
  for (int off = 16; off; off >>= 1) ss += __shfl_xor(ss, off, 32);
  const float scale = 1.0f / fmaxf(sqrtf(ss), 1e-12f);
  ushort4 o;
  o.x = f2bf(v.x * scale);
  o.y = f2bf(v.y * scale);
  o.z = f2bf(v.z * scale);
  o.w = f2bf(v.w * scale);
  reinterpret_cast<ushort4*>(xn)[row * 32 + lane] = o;
}

// --------------------- Kernel 2: fused GEMM + top-k ------------------------
// Block = 16-row stripe, 8 waves. Wave w processes column tiles w, w+8, ...
// Per tile: issue adjacency NT clause first (max outstanding HBM stream),
// then one 8x b128 clause for all four B fragments, then 4 back-to-back
// v_wmma_f32_16x16x32_bf16 (K=128), then the elementwise epilogue.
__global__ void __launch_bounds__(256)
ans_fused_kernel(const float* __restrict__ adj, const unsigned short* __restrict__ xn,
                 float* __restrict__ out_vals, float* __restrict__ out_idx) {
  __shared__ float sc[8][16][17];      // per-wave score staging (padded)
  __shared__ float candV[16][256];     // merge candidates: [row][wave*2+hi][16]
  __shared__ int   candI[16][256];
  __shared__ float rowsum[16];

  const int tid  = threadIdx.x;
  const int w    = tid >> 5;
  const int lane = tid & 31;
  const int cl   = lane & 15;          // column-in-tile (phase A) / row (phase B)
  const int hi   = lane >> 4;
  const int i0   = blockIdx.x * 16;

  if (tid < 16) rowsum[tid] = 0.0f;
  __syncthreads();

  const uint4* xv = reinterpret_cast<const uint4*>(xn);  // 16 uint4 per row

  // A fragments: rows i0..i0+15, all K=128 (resident for the whole column walk).
  // 16-bit A layout: lanes 0-15 hold M=lane, K {k0..k0+7, k0+16..k0+23};
  // lanes 16-31 hold K {k0+8..k0+15, k0+24..k0+31}.
  v16bf afrag[4];
  {
    const int i = i0 + cl;
#pragma unroll
    for (int kc = 0; kc < 4; ++kc) {
      uint4 a = xv[i * 16 + kc * 4 + hi];
      uint4 b = xv[i * 16 + kc * 4 + 2 + hi];
      afrag[kc] = as_v16bf(a, b);
    }
  }

  const float NEG_INF = -__builtin_inff();
  const float POS_INF =  __builtin_inff();

  float tv[KSEL], bv[KSEL];
  int   ti[KSEL], bi[KSEL];
#pragma unroll
  for (int s = 0; s < KSEL; ++s) { tv[s] = NEG_INF; ti[s] = -1; bv[s] = POS_INF; bi[s] = -1; }

  float rs_acc[8];
#pragma unroll
  for (int v = 0; v < 8; ++v) rs_acc[v] = 0.0f;

  for (int t = 0; t < 64; ++t) {
    const int j0 = (w + t * 8) * 16;
    const int j  = j0 + cl;

    // (1) Adjacency stream first: 8 NT b32 loads (non-temporal keeps the
    // 256 MB single-pass stream out of L2 so xn stays resident). Issuing the
    // clause before the B loads / WMMAs maximizes outstanding HBM requests.
    const float* arow = adj + (size_t)(i0 + 8 * hi) * N_NODES + (j0 + cl);
    float av[8];
#pragma unroll
    for (int v = 0; v < 8; ++v)
      av[v] = __builtin_nontemporal_load(arow + (size_t)v * N_NODES);

    // (2) All four B fragments up-front (one 8x b128 clause, one wait).
    // 16-bit B layout: lanes 0-15 = col N=lane, K {k0..k0+7, k0+16..k0+23};
    // lanes 16-31 = same cols, K {k0+8..k0+15, k0+24..k0+31}. xn is 2 MB: L2 hits.
    v16bf bfrag[4];
#pragma unroll
    for (int kc = 0; kc < 4; ++kc) {
      uint4 a = xv[j * 16 + kc * 4 + hi];
      uint4 b = xv[j * 16 + kc * 4 + 2 + hi];
      bfrag[kc] = as_v16bf(a, b);
    }

    // (3) sim tile (16x16, rows i0.., cols j0..): 4 chained bf16 WMMAs.
    v8f c = {};
#pragma unroll
    for (int kc = 0; kc < 4; ++kc)
      c = __builtin_amdgcn_wmma_f32_16x16x32_bf16(
              false, afrag[kc], false, bfrag[kc], (short)0, c, false, false);

    // (4) C layout: lane owns column j0+cl; VGPR v = row (v + 8*hi).
    // Weight, mask, accumulate row sums, stage masked scores to LDS.
#pragma unroll
    for (int v = 0; v < 8; ++v) {
      float we = c[v] * av[v];             // non-neighbors contribute 0 to rowsum
      rs_acc[v] += we;
      sc[w][v + 8 * hi][cl] = (av[v] > 0.0f) ? we : NEG_INF;
    }

    // (5) Phase B (same wave, LDS in-order): lane owns row=cl, half=hi.
    for (int u = 0; u < 8; ++u) {
      const int   col = hi * 8 + u;
      const float s   = sc[w][cl][col];
      const int   idx = j0 + col;
      if (s > tv[KSEL - 1]) ins_desc(tv, ti, s, idx);          // top-16 of weighted
      const float sb = (s == NEG_INF) ? POS_INF : s;           // exclude masked
      if (sb < bv[KSEL - 1]) ins_asc(bv, bi, sb, idx);         // bottom-16 (rowsum<0)
    }
  }

  // Row-sum reduction across the 16 column-lanes of every wave.
#pragma unroll
  for (int v = 0; v < 8; ++v) atomicAdd(&rowsum[v + 8 * hi], rs_acc[v]);
  __syncthreads();

  // Dividing by rowsum preserves order iff rowsum >= 0; pick list by sign.
  {
    const float rs     = rowsum[cl];
    const bool  useTop = (rs >= 0.0f);
    const int   base   = (w * 2 + hi) * 16;
#pragma unroll
    for (int s = 0; s < KSEL; ++s) {
      candV[cl][base + s] = useTop ? tv[s] : bv[s];
      candI[cl][base + s] = useTop ? ti[s] : bi[s];
    }
  }
  __syncthreads();

  // Final selection: wave w handles rows 2w, 2w+1; 16 rounds of wave argmax/argmin.
  for (int rr = 0; rr < 2; ++rr) {
    const int   row    = w * 2 + rr;
    const float rs     = rowsum[row];
    const bool  maxdir = (rs >= 0.0f);
    float cv[8];
#pragma unroll
    for (int u = 0; u < 8; ++u) cv[u] = candV[row][lane * 8 + u];

    for (int q = 0; q < KSEL; ++q) {
      float b  = cv[0];
      int   bp = 0;
#pragma unroll
      for (int u = 1; u < 8; ++u) {
        const bool take = maxdir ? (cv[u] > b) : (cv[u] < b);
        if (take) { b = cv[u]; bp = u; }
      }
      int bg = lane * 8 + bp;
#pragma unroll
      for (int off = 16; off; off >>= 1) {
        const float ob = __shfl_xor(b, off, 32);
        const int   og = __shfl_xor(bg, off, 32);
        bool take = maxdir ? (ob > b) : (ob < b);
        take = take || (ob == b && og < bg);
        if (take) { b = ob; bg = og; }
      }
      // knock out the winner (unrolled: no dynamic register indexing)
      const bool mine = (lane == (bg >> 3));
      const int  lp   = bg & 7;
#pragma unroll
      for (int u = 0; u < 8; ++u)
        if (mine && u == lp) cv[u] = maxdir ? NEG_INF : POS_INF;

      if (lane == 0) {
        const int gi = i0 + row;
        float pv = (b == NEG_INF || b == POS_INF) ? NEG_INF : (b / rs);
        out_vals[gi * KSEL + q] = pv;
        out_idx [gi * KSEL + q] = (float)candI[row][bg];
      }
    }
  }
}

// ---------------------------------------------------------------------------
extern "C" void kernel_launch(void* const* d_in, const int* in_sizes, int n_in,
                              void* d_out, int out_size, void* d_ws, size_t ws_size,
                              hipStream_t stream) {
  (void)in_sizes; (void)n_in; (void)out_size; (void)ws_size;
  const float* adj = (const float*)d_in[0];   // [N,N] f32
  const float* x   = (const float*)d_in[1];   // [N,D] f32
  // d_in[2] = labels (unused), d_in[3] = k (==16, baked in)

  unsigned short* xn = (unsigned short*)d_ws; // [N,D] bf16, 2 MB scratch

  float* out_vals = (float*)d_out;                      // [N,16]
  float* out_idx  = (float*)d_out + N_NODES * KSEL;     // [N,16] indices as float

  l2norm_bf16_kernel<<<N_NODES / 8, 256, 0, stream>>>(x, xn);
  ans_fused_kernel<<<N_NODES / 16, 256, 0, stream>>>(adj, xn, out_vals, out_idx);
}
